// MultiHeadAttention_27745488732960
// MI455X (gfx1250) — matrix operand order repaired and checked by reading, not compiled, not run on hardware
//
#include <hip/hip_runtime.h>
#include <hip/hip_fp16.h>

typedef __attribute__((ext_vector_type(16))) _Float16 v16h;
typedef __attribute__((ext_vector_type(8)))  _Float16 v8h;
typedef __attribute__((ext_vector_type(8)))  float    v8f;
typedef __attribute__((ext_vector_type(4)))  float    v4f;
typedef __attribute__((ext_vector_type(4)))  unsigned int u32x4;
typedef __attribute__((ext_vector_type(8)))  int      i32x8;
typedef __attribute__((ext_vector_type(4)))  int      i32x4;
// GCC-style vector matching the async-to-LDS builtin's pointer pointee type.
typedef int athena_v4i __attribute__((vector_size(16)));

// ---- CDNA5 async / TDM feature detection (compile-safe) --------------------
#if defined(__has_builtin)
#if __has_builtin(__builtin_amdgcn_tensor_load_to_lds) && \
    __has_builtin(__builtin_amdgcn_s_wait_tensorcnt)
#define ATHENA_HAS_TDM 1
#endif
#if __has_builtin(__builtin_amdgcn_global_load_async_to_lds_b128) && \
    __has_builtin(__builtin_amdgcn_s_wait_asynccnt)
#define ATHENA_HAS_ASYNC 1
#endif
#endif

// Both pointer params of global_load_async_to_lds_b128 are generic v4i*.
#define ATHENA_GPTR(p) ((athena_v4i*)(p))
#define ATHENA_LPTR(p) ((athena_v4i*)(p))

namespace {

constexpr int kDModel = 1024;
constexpr int kHeads  = 16;
constexpr int kDk     = 64;
constexpr int kBatch  = 2;
constexpr int kSeq    = 2048;
constexpr int kM      = kBatch * kSeq;  // 4096

__device__ __forceinline__ v16h mkv16(v8h lo, v8h hi) {
  v16h r;
#pragma unroll
  for (int i = 0; i < 8; ++i) { r[i] = lo[i]; r[i + 8] = hi[i]; }
  return r;
}

__device__ __forceinline__ v8f wmma_f16(v16h a, v16h b, v8f c) {
  return __builtin_amdgcn_wmma_f32_16x16x32_f16(false, a, false, b,
                                                (short)0, c, false, false);
}

#if defined(ATHENA_HAS_TDM)
// One TDM descriptor moves a 128-row x 32-half f16 tile (row stride = K halves)
// into LDS with 4-dword padding after every 16 dwords (i.e. LDS row stride of
// 40 halves), matching the WMMA staging layout. Issued by wave 0 only.
__device__ __forceinline__ void tdm_load_tile_f16(const _Float16* gsrc,
                                                  unsigned int lds_off,
                                                  int row_stride, int tensor_rows) {
  unsigned long long ga = (unsigned long long)(uintptr_t)gsrc;
  u32x4 g0;
  g0[0] = 1u;                                   // count=1 user descriptor
  g0[1] = lds_off;                              // lds_addr (bytes)
  g0[2] = (unsigned int)(ga & 0xffffffffull);   // global_addr[31:0]
  g0[3] = (unsigned int)((ga >> 32) & 0x1ffffffull) | 0x80000000u;  // addr[56:32] | type=2
  const int td0 = row_stride;                   // tensor dim0 (halves)
  const int td1 = tensor_rows;                  // tensor dim1 (rows)
  i32x8 g1;
  g1[0] = (1 << 16) |                           // data_size = 2 bytes
          (1 << 20) |                           // pad_enable
          (3 << 22) |                           // pad_interval: 16 dwords
          (3 << 25);                            // pad_amount:   4 dwords
  g1[1] = (td0 & 0xffff) << 16;                               // tensor_dim0 lo
  g1[2] = ((td0 >> 16) & 0xffff) | ((td1 & 0xffff) << 16);    // dim0 hi | dim1 lo
  g1[3] = ((td1 >> 16) & 0xffff) | (32 << 16);                // dim1 hi | tile_dim0=32
  g1[4] = 128;                                                // tile_dim1=128, tile_dim2=0
  g1[5] = row_stride;                                         // tensor_dim0_stride lo
  g1[6] = 0;
  g1[7] = 0;
  i32x4 z4 = {0, 0, 0, 0};
  i32x8 z8 = {0, 0, 0, 0, 0, 0, 0, 0};
  __builtin_amdgcn_tensor_load_to_lds(g0, g1, z4, z4, z8, 0);
}
#endif

// ---------------------------------------------------------------------------
// fp32 -> f16 elementwise conversion (8 elements / thread, sizes % 2048 == 0)
// ---------------------------------------------------------------------------
__global__ __launch_bounds__(256) void f32_to_f16(const float* __restrict__ src,
                                                  _Float16* __restrict__ dst, int n) {
  const int i = (blockIdx.x * 256 + threadIdx.x) * 8;
  if (i + 8 <= n) {
    v4f a0 = *(const v4f*)(src + i);
    v4f a1 = *(const v4f*)(src + i + 4);
    v8h h;
#pragma unroll
    for (int j = 0; j < 4; ++j) { h[j] = (_Float16)a0[j]; h[j + 4] = (_Float16)a1[j]; }
    *(v8h*)(dst + i) = h;
  }
}

// ---------------------------------------------------------------------------
// Y[m,n] = sum_k A[m,k] * W[n,k] + bias[n]   (A @ W^T + b), all-f16 operands,
// fp32 accumulation. Block tile 128x128, BK=32, 256 threads = 8 waves (32x64
// per wave). Double-buffered LDS; tiles staged by TDM / async-to-LDS / regs.
// ---------------------------------------------------------------------------
template <typename TOUT>
__global__ __launch_bounds__(256) void gemm_wmma(
    const _Float16* __restrict__ A, const _Float16* __restrict__ W,
    const float* __restrict__ bias, TOUT* __restrict__ Y,
    int M, int N, int K) {
  constexpr int BM = 128, BN = 128, BK = 32, LDT = BK + 8;
  constexpr int TILE = BM * LDT;  // halves per buffer
  __shared__ __align__(16) _Float16 As[2 * TILE];
  __shared__ __align__(16) _Float16 Ws[2 * TILE];

  const int tid  = threadIdx.x;
  const int lane = tid & 31;
  const int wave = tid >> 5;
  const int wm   = wave & 3;
  const int wn   = wave >> 2;
  const int half = lane >> 4;
  const int l16  = lane & 15;
  const int bm   = blockIdx.x * BM;
  const int bn   = blockIdx.y * BN;
  const int lrow = tid >> 1;        // loader row 0..127
  const int lcb  = (tid & 1) * 16;  // loader col base 0/16

  v8f acc[2][4];
#pragma unroll
  for (int mi = 0; mi < 2; ++mi)
#pragma unroll
    for (int ni = 0; ni < 4; ++ni) acc[mi][ni] = (v8f)(0.0f);

  const int NK = K / BK;

  // ---- tile staging (issue side) ----
#if defined(ATHENA_HAS_TDM)
#define STAGE_ISSUE(KT, BUF)                                                      \
  do {                                                                            \
    if (wave == 0) {                                                              \
      tdm_load_tile_f16(A + (size_t)bm * K + (KT) * BK,                           \
                        (unsigned int)(uintptr_t)&As[(BUF) * TILE], K, M - bm);   \
      tdm_load_tile_f16(W + (size_t)bn * K + (KT) * BK,                           \
                        (unsigned int)(uintptr_t)&Ws[(BUF) * TILE], K, N - bn);   \
    }                                                                             \
  } while (0)
#define STAGE_WAIT()                                                              \
  do { if (wave == 0) __builtin_amdgcn_s_wait_tensorcnt(0); } while (0)
#elif defined(ATHENA_HAS_ASYNC)
#define STAGE_ISSUE(KT, BUF)                                                      \
  do {                                                                            \
    const _Float16* as_ = A + (size_t)(bm + lrow) * K + (KT) * BK + lcb;          \
    const _Float16* ws_ = W + (size_t)(bn + lrow) * K + (KT) * BK + lcb;          \
    _Float16* ad_ = &As[(BUF) * TILE + lrow * LDT + lcb];                         \
    _Float16* wd_ = &Ws[(BUF) * TILE + lrow * LDT + lcb];                         \
    __builtin_amdgcn_global_load_async_to_lds_b128(ATHENA_GPTR(as_), ATHENA_LPTR(ad_), 0, 0);     \
    __builtin_amdgcn_global_load_async_to_lds_b128(ATHENA_GPTR(as_ + 8), ATHENA_LPTR(ad_ + 8), 0, 0); \
    __builtin_amdgcn_global_load_async_to_lds_b128(ATHENA_GPTR(ws_), ATHENA_LPTR(wd_), 0, 0);     \
    __builtin_amdgcn_global_load_async_to_lds_b128(ATHENA_GPTR(ws_ + 8), ATHENA_LPTR(wd_ + 8), 0, 0); \
  } while (0)
#define STAGE_WAIT() __builtin_amdgcn_s_wait_asynccnt(0)
#endif

#if defined(ATHENA_HAS_TDM) || defined(ATHENA_HAS_ASYNC)
  STAGE_ISSUE(0, 0);
  STAGE_WAIT();
#else
  {  // synchronous prologue
    const _Float16* as_ = A + (size_t)(bm + lrow) * K + lcb;
    const _Float16* ws_ = W + (size_t)(bn + lrow) * K + lcb;
#pragma unroll
    for (int i = 0; i < 2; ++i) {
      *(v8h*)&As[lrow * LDT + lcb + i * 8] = *(const v8h*)(as_ + i * 8);
      *(v8h*)&Ws[lrow * LDT + lcb + i * 8] = *(const v8h*)(ws_ + i * 8);
    }
  }
#endif
  __syncthreads();

  for (int kt = 0; kt < NK; ++kt) {
    const int cur = kt & 1;
    const int nxt = cur ^ 1;
    const bool more = (kt + 1) < NK;

#if defined(ATHENA_HAS_TDM) || defined(ATHENA_HAS_ASYNC)
    if (more) STAGE_ISSUE(kt + 1, nxt);
#else
    v8h ra[2], rw[2];
    if (more) {
      const _Float16* as_ = A + (size_t)(bm + lrow) * K + (kt + 1) * BK + lcb;
      const _Float16* ws_ = W + (size_t)(bn + lrow) * K + (kt + 1) * BK + lcb;
#pragma unroll
      for (int i = 0; i < 2; ++i) {
        ra[i] = *(const v8h*)(as_ + i * 8);
        rw[i] = *(const v8h*)(ws_ + i * 8);
      }
    }
#endif

    // ---- compute on current buffer: 2x4 WMMA subtiles ----
    const _Float16* Ab = &As[cur * TILE];
    const _Float16* Wb = &Ws[cur * TILE];
    v16h afrag[2];
#pragma unroll
    for (int mi = 0; mi < 2; ++mi) {
      const _Float16* p = &Ab[(wm * 32 + mi * 16 + l16) * LDT];
      afrag[mi] = mkv16(*(const v8h*)(p + half * 8),
                        *(const v8h*)(p + 16 + half * 8));
    }
#pragma unroll
    for (int ni = 0; ni < 4; ++ni) {
      const _Float16* p = &Wb[(wn * 64 + ni * 16 + l16) * LDT + half * 8];
      v16h bfrag = mkv16(*(const v8h*)p, *(const v8h*)(p + 16));
#pragma unroll
      for (int mi = 0; mi < 2; ++mi)
        acc[mi][ni] = wmma_f16(afrag[mi], bfrag, acc[mi][ni]);
    }

#if defined(ATHENA_HAS_TDM) || defined(ATHENA_HAS_ASYNC)
    if (more) STAGE_WAIT();
#else
    if (more) {
#pragma unroll
      for (int i = 0; i < 2; ++i) {
        *(v8h*)&As[nxt * TILE + lrow * LDT + lcb + i * 8] = ra[i];
        *(v8h*)&Ws[nxt * TILE + lrow * LDT + lcb + i * 8] = rw[i];
      }
    }
#endif
    __syncthreads();
  }
#undef STAGE_ISSUE
#undef STAGE_WAIT

  // ---- epilogue: bias + store (D layout: VGPR r, lane -> M=r+8*half, N=l16)
#pragma unroll
  for (int ni = 0; ni < 4; ++ni) {
    const int n = bn + wn * 64 + ni * 16 + l16;
    const float bv = bias[n];
#pragma unroll
    for (int mi = 0; mi < 2; ++mi) {
      const int mbase = bm + wm * 32 + mi * 16 + half * 8;
#pragma unroll
      for (int r = 0; r < 8; ++r)
        Y[(size_t)(mbase + r) * N + n] = (TOUT)(acc[mi][ni][r] + bv);
    }
  }
}

// ---------------------------------------------------------------------------
// Flash attention (causal, online softmax). One block per (batch*head,
// 64-query-row tile), 128 threads = 4 waves, 16 query rows per wave.
// ---------------------------------------------------------------------------
__global__ __launch_bounds__(128) void flash_attn(
    const _Float16* __restrict__ q16, const _Float16* __restrict__ k16,
    const _Float16* __restrict__ v16, _Float16* __restrict__ attn) {
  constexpr int KLD = kDk + 8;
  constexpr int VLD = 64 + 8;
  constexpr int PLD = 64 + 8;
  __shared__ __align__(16) _Float16 Ks[64 * KLD];      // [key][d]
  __shared__ __align__(16) _Float16 Vt[kDk * VLD];     // [d][key] transposed
  __shared__ __align__(16) _Float16 Ps[4 * 16 * PLD];  // per-wave P tile

  const int tid  = threadIdx.x;
  const int lane = tid & 31;
  const int wave = tid >> 5;
  const int half = lane >> 4;
  const int l16  = lane & 15;

  const int qb = blockIdx.x;
  const int bh = blockIdx.y;
  const int b  = bh / kHeads;
  const int h  = bh % kHeads;
  const int bS = b * kSeq;

  const int qrow = bS + qb * 64 + wave * 16 + l16;
  const _Float16* qp = q16 + (size_t)qrow * kDModel + h * kDk;
  v16h qfrag[2];
#pragma unroll
  for (int kt = 0; kt < 2; ++kt)
    qfrag[kt] = mkv16(*(const v8h*)(qp + kt * 32 + half * 8),
                      *(const v8h*)(qp + kt * 32 + 16 + half * 8));

  v8f oacc[4];
#pragma unroll
  for (int di = 0; di < 4; ++di) oacc[di] = (v8f)(0.0f);
  float mstate[8], lstate[8];
#pragma unroll
  for (int r = 0; r < 8; ++r) { mstate[r] = -3.0e38f; lstate[r] = 0.0f; }

  const float scale = 0.125f;  // 1/sqrt(64)

  for (int kb = 0; kb <= qb; ++kb) {
    // ---- stage K (natural, async when available) + V (manual transpose) ----
    {
      const int r  = tid >> 1;
      const int cp = (tid & 1) * 32;
      const size_t grow = (size_t)(bS + kb * 64 + r) * kDModel + h * kDk + cp;
      const _Float16* ksrc = k16 + grow;
      _Float16* kdst = &Ks[r * KLD + cp];
#if defined(ATHENA_HAS_ASYNC)
      __builtin_amdgcn_global_load_async_to_lds_b128(ATHENA_GPTR(ksrc), ATHENA_LPTR(kdst), 0, 0);
      __builtin_amdgcn_global_load_async_to_lds_b128(ATHENA_GPTR(ksrc + 8), ATHENA_LPTR(kdst + 8), 0, 0);
      __builtin_amdgcn_global_load_async_to_lds_b128(ATHENA_GPTR(ksrc + 16), ATHENA_LPTR(kdst + 16), 0, 0);
      __builtin_amdgcn_global_load_async_to_lds_b128(ATHENA_GPTR(ksrc + 24), ATHENA_LPTR(kdst + 24), 0, 0);
#else
#pragma unroll
      for (int i = 0; i < 4; ++i)
        *(v8h*)(kdst + i * 8) = *(const v8h*)(ksrc + i * 8);
#endif
#pragma unroll
      for (int i = 0; i < 4; ++i) {
        v8h vv = *(const v8h*)(v16 + grow + i * 8);
#pragma unroll
        for (int j = 0; j < 8; ++j) Vt[(cp + i * 8 + j) * VLD + r] = vv[j];
      }
#if defined(ATHENA_HAS_ASYNC)
      __builtin_amdgcn_s_wait_asynccnt(0);
#endif
    }
    __syncthreads();

    // ---- S = Q K^T (16x64 per wave) ----
    v8f sacc[4];
#pragma unroll
    for (int ni = 0; ni < 4; ++ni) sacc[ni] = (v8f)(0.0f);
#pragma unroll
    for (int kt = 0; kt < 2; ++kt)
#pragma unroll
      for (int ni = 0; ni < 4; ++ni) {
        const _Float16* p = &Ks[(ni * 16 + l16) * KLD + kt * 32 + half * 8];
        v16h bfrag = mkv16(*(const v8h*)p, *(const v8h*)(p + 16));
        sacc[ni] = wmma_f16(qfrag[kt], bfrag, sacc[ni]);
      }

    // ---- causal mask + online softmax ----
    const int qa = qb * 64 + wave * 16 + half * 8;
    const int ka = kb * 64 + l16;
    float pexp[4][8];
#pragma unroll
    for (int r = 0; r < 8; ++r) {
      float sv[4];
#pragma unroll
      for (int ni = 0; ni < 4; ++ni) {
        float x = sacc[ni][r] * scale;
        if (kb == qb && (ka + ni * 16) > (qa + r)) x = -3.0e38f;
        sv[ni] = x;
      }
      float mrow = fmaxf(fmaxf(sv[0], sv[1]), fmaxf(sv[2], sv[3]));
#pragma unroll
      for (int off = 1; off < 16; off <<= 1)
        mrow = fmaxf(mrow, __shfl_xor(mrow, off, 32));
      const float mnew = fmaxf(mstate[r], mrow);
      const float corr = __expf(mstate[r] - mnew);
      float rsum = 0.0f;
#pragma unroll
      for (int ni = 0; ni < 4; ++ni) {
        const float p = __expf(sv[ni] - mnew);
        pexp[ni][r] = p;
        rsum += p;
      }
#pragma unroll
      for (int off = 1; off < 16; off <<= 1) rsum += __shfl_xor(rsum, off, 32);
      lstate[r] = lstate[r] * corr + rsum;
      mstate[r] = mnew;
#pragma unroll
      for (int di = 0; di < 4; ++di) oacc[di][r] *= corr;
    }

    // ---- D-fragment -> A-fragment relayout of P through per-wave LDS ----
    _Float16* pw = &Ps[wave * 16 * PLD];
#pragma unroll
    for (int r = 0; r < 8; ++r)
#pragma unroll
      for (int ni = 0; ni < 4; ++ni)
        pw[(r + 8 * half) * PLD + ni * 16 + l16] = (_Float16)pexp[ni][r];

    // ---- O += P V ----
#pragma unroll
    for (int kt = 0; kt < 2; ++kt) {
      const _Float16* ap = &pw[l16 * PLD + kt * 32];
      v16h afrag = mkv16(*(const v8h*)(ap + half * 8),
                         *(const v8h*)(ap + 16 + half * 8));
#pragma unroll
      for (int di = 0; di < 4; ++di) {
        const _Float16* bp = &Vt[(di * 16 + l16) * VLD + kt * 32 + half * 8];
        v16h bfrag = mkv16(*(const v8h*)bp, *(const v8h*)(bp + 16));
        oacc[di] = wmma_f16(afrag, bfrag, oacc[di]);
      }
    }
    __syncthreads();
  }

  // ---- normalize + store f16 ----
#pragma unroll
  for (int r = 0; r < 8; ++r) {
    const float inv = 1.0f / lstate[r];
    const int row = bS + qb * 64 + wave * 16 + r + 8 * half;
    const size_t base = (size_t)row * kDModel + h * kDk;
#pragma unroll
    for (int di = 0; di < 4; ++di)
      attn[base + di * 16 + l16] = (_Float16)(oacc[di][r] * inv);
  }
}

}  // namespace

extern "C" void kernel_launch(void* const* d_in, const int* in_sizes, int n_in,
                              void* d_out, int out_size, void* d_ws, size_t ws_size,
                              hipStream_t stream) {
  (void)in_sizes; (void)n_in; (void)out_size; (void)ws_size;
  const float* query = (const float*)d_in[0];
  const float* keyi  = (const float*)d_in[1];
  const float* value = (const float*)d_in[2];
  // d_in[3] = causal mask (tril bool) -- handled analytically in flash_attn
  const float* W_q = (const float*)d_in[4];
  const float* b_q = (const float*)d_in[5];
  const float* W_k = (const float*)d_in[6];
  const float* b_k = (const float*)d_in[7];
  const float* W_v = (const float*)d_in[8];
  const float* b_v = (const float*)d_in[9];
  const float* W_o = (const float*)d_in[10];
  const float* b_o = (const float*)d_in[11];
  float* out = (float*)d_out;

  const size_t mat = (size_t)kM * kDModel;        // 4M elements
  const size_t wmat = (size_t)kDModel * kDModel;  // 1M elements
  _Float16* p = (_Float16*)d_ws;
  _Float16* qx16 = p; p += mat;   // converted activations
  _Float16* kx16 = p; p += mat;
  _Float16* vx16 = p; p += mat;
  _Float16* wq16 = p; p += wmat;  // converted weights
  _Float16* wk16 = p; p += wmat;
  _Float16* wv16 = p; p += wmat;
  _Float16* wo16 = p; p += wmat;
  _Float16* q16  = p; p += mat;   // projection outputs
  _Float16* k16  = p; p += mat;
  _Float16* v16  = p; p += mat;
  _Float16* a16  = p; p += mat;   // attention output

  // ---- fp32 -> f16 conversion passes ----
  f32_to_f16<<<dim3((unsigned)(mat / 2048)), 256, 0, stream>>>(query, qx16, (int)mat);
  f32_to_f16<<<dim3((unsigned)(mat / 2048)), 256, 0, stream>>>(keyi,  kx16, (int)mat);
  f32_to_f16<<<dim3((unsigned)(mat / 2048)), 256, 0, stream>>>(value, vx16, (int)mat);
  f32_to_f16<<<dim3((unsigned)(wmat / 2048)), 256, 0, stream>>>(W_q, wq16, (int)wmat);
  f32_to_f16<<<dim3((unsigned)(wmat / 2048)), 256, 0, stream>>>(W_k, wk16, (int)wmat);
  f32_to_f16<<<dim3((unsigned)(wmat / 2048)), 256, 0, stream>>>(W_v, wv16, (int)wmat);
  f32_to_f16<<<dim3((unsigned)(wmat / 2048)), 256, 0, stream>>>(W_o, wo16, (int)wmat);

  // ---- WMMA projections ----
  dim3 gGrid(kM / 128, kDModel / 128);
  gemm_wmma<_Float16><<<gGrid, 256, 0, stream>>>(qx16, wq16, b_q, q16, kM, kDModel, kDModel);
  gemm_wmma<_Float16><<<gGrid, 256, 0, stream>>>(kx16, wk16, b_k, k16, kM, kDModel, kDModel);
  gemm_wmma<_Float16><<<gGrid, 256, 0, stream>>>(vx16, wv16, b_v, v16, kM, kDModel, kDModel);

  // ---- fused causal flash attention ----
  dim3 aGrid(kSeq / 64, kBatch * kHeads);
  flash_attn<<<aGrid, 128, 0, stream>>>(q16, k16, v16, a16);

  // ---- output projection (fp32 result) ----
  gemm_wmma<float><<<gGrid, 256, 0, stream>>>(a16, wo16, b_o, out, kM, kDModel, kDModel);
}
